// MemoryRNN_23897198035120
// MI455X (gfx1250) — matrix operand order repaired and checked by reading, not compiled, not run on hardware
//
#include <hip/hip_runtime.h>
#include <hip/hip_bf16.h>

typedef __attribute__((ext_vector_type(16))) __bf16 v16bf;
typedef __attribute__((ext_vector_type(8)))  __bf16 v8bf;
typedef __attribute__((ext_vector_type(4)))  __bf16 v4bf;
typedef __attribute__((ext_vector_type(8)))  float  v8f;

#define T_STEPS 512
#define BB 51
#define DD 4096
#define HH 1024
#define RR 51
#define MTOT (T_STEPS * BB)   /* 26112 */
#define N6H (6 * HH)          /* 6144  */
#define N5H (5 * HH)          /* 5120  */
#define LDSK (HH + 8)         /* padded LDS row (bf16 elems): 2064B -> bank spread */
#define NBLK 64               /* persistent recurrence blocks */

// ---------- fragment helpers ----------
// A (16x32 bf16): lane holds row ln; K chunk [base..base+15],
// base = hi8?16:0 (caller pre-offsets pointer by base).
static __device__ __forceinline__ v16bf load_a_bf16(const __bf16* __restrict__ p) {
  v8bf lo = *(const v8bf*)(p);
  v8bf hi = *(const v8bf*)(p + 8);
  v16bf a;
#pragma unroll
  for (int i = 0; i < 8; ++i) { a[i] = lo[i]; a[8 + i] = hi[i]; }
  return a;
}

// B (32x16 bf16) from K-contiguous row: lanes 0-15 K=0..15, lanes 16-31 K=16..31
// (caller pre-offsets pointer by hi8?16:0).
static __device__ __forceinline__ v16bf load_b_bf16(const __bf16* __restrict__ p) {
  v8bf lo = *(const v8bf*)(p);
  v8bf hi = *(const v8bf*)(p + 8);
  v16bf b;
#pragma unroll
  for (int i = 0; i < 8; ++i) { b[i] = lo[i]; b[8 + i] = hi[i]; }
  return b;
}

static __device__ __forceinline__ float sigmoidf_(float x) {
  return 1.0f / (1.0f + __expf(-x));
}

// ---------- prep kernels ----------
__global__ void k_transpose_cvt(const float* __restrict__ src, __bf16* __restrict__ dst,
                                int K, int N, long long total /* Npad*K */) {
  long long stride = (long long)gridDim.x * blockDim.x;
  for (long long idx = (long long)blockIdx.x * blockDim.x + threadIdx.x; idx < total; idx += stride) {
    int n = (int)(idx / K);
    int k = (int)(idx - (long long)n * K);
    dst[idx] = (n < N) ? (__bf16)src[(size_t)k * N + n] : (__bf16)0.0f;
  }
}

__global__ void k_cvt_bf16_v4(const float* __restrict__ src, __bf16* __restrict__ dst,
                              long long n4) {
  long long stride = (long long)gridDim.x * blockDim.x;
  for (long long i = (long long)blockIdx.x * blockDim.x + threadIdx.x; i < n4; i += stride) {
    float4 v = ((const float4*)src)[i];
    v4bf o; o[0] = (__bf16)v.x; o[1] = (__bf16)v.y; o[2] = (__bf16)v.z; o[3] = (__bf16)v.w;
    ((v4bf*)dst)[i] = o;
  }
}

__global__ void k_copy_f32(const float* __restrict__ src, float* __restrict__ dst, int n) {
  int stride = gridDim.x * blockDim.x;
  for (int i = blockIdx.x * blockDim.x + threadIdx.x; i < n; i += stride) dst[i] = src[i];
}

__global__ void k_zero_i32(int* p, int n) {
  int i = blockIdx.x * blockDim.x + threadIdx.x;
  if (i < n) p[i] = 0;
}

// ---------- big GEMM: pin_all[M,6H] = seqBF[M,D] @ WinT^T + b_in ----------
// 4 waves/block; wave w: rows m0 = by*64 + w*16, cols n0 = bx*128 (8 subtiles of 16).
__global__ __launch_bounds__(128) void k_gemm_pin(
    const __bf16* __restrict__ A,     // [MTOT, DD] bf16
    const __bf16* __restrict__ WT,    // [N6H, DD] bf16 (K-contiguous)
    const float* __restrict__ bias,   // [N6H]
    float* __restrict__ C)            // [MTOT, N6H]
{
  const int lane = threadIdx.x & 31;
  const int wave = threadIdx.x >> 5;
  const int ln = lane & 15;
  const int hi8 = lane >> 4;
  const int m0 = blockIdx.y * 64 + wave * 16;
  const int n0 = blockIdx.x * 128;

  const __bf16* arow = A + (size_t)(m0 + ln) * DD + (hi8 ? 16 : 0);
  const int kb_b = hi8 ? 16 : 0;

  v8f acc[8];
#pragma unroll
  for (int j = 0; j < 8; ++j)
#pragma unroll
    for (int i = 0; i < 8; ++i) acc[j][i] = 0.0f;

  for (int k0 = 0; k0 < DD; k0 += 32) {
    const __bf16* ap = arow + k0;
    __builtin_prefetch(ap + 128, 0, 1);
    v16bf afrag = load_a_bf16(ap);
#pragma unroll
    for (int j = 0; j < 8; ++j) {
      const __bf16* wp = WT + (size_t)(n0 + j * 16 + ln) * DD + k0 + kb_b;
      v16bf bfrag = load_b_bf16(wp);
      acc[j] = __builtin_amdgcn_wmma_f32_16x16x32_bf16(
          false, afrag, false, bfrag, (short)0, acc[j], false, false);
    }
  }
#pragma unroll
  for (int j = 0; j < 8; ++j) {
    const int n = n0 + j * 16 + ln;
    const float bb = bias[n];
#pragma unroll
    for (int i = 0; i < 8; ++i) {
      const int m = m0 + i + 8 * hi8;
      __builtin_nontemporal_store(acc[j][i] + bb, &C[(size_t)m * N6H + n]);  // streamed, read once
    }
  }
}

// ---------- persistent recurrence ----------
// 64 blocks x 128 threads. Block = one 16-col n-tile; waves = 4 m-tiles.
// W_s slice (5 gates x 16 cols x K) staged once into LDS via async-to-LDS,
// then 512 steps run back-to-back with a software grid barrier.
__global__ __launch_bounds__(128) void k_recurrence(
    const float* __restrict__ pin_all,   // [MTOT, 6H]
    const __bf16* __restrict__ WsT,      // [5H, H] bf16 (K-contiguous)
    const float* __restrict__ b_s,       // [5H]
    __bf16* __restrict__ hbuf0, float* __restrict__ cbuf0,   // ping-pong state
    __bf16* __restrict__ hbuf1, float* __restrict__ cbuf1,
    __bf16* __restrict__ HallBF,         // [MTOT, H] bf16
    const int* __restrict__ labels,      // [MTOT]
    int* __restrict__ bar)
{
  __shared__ __bf16 ldsB[5 * 16 * LDSK];   // 165,120 B of the 320KB WGP LDS

  const int lane = threadIdx.x & 31;
  const int wave = threadIdx.x >> 5;       // m_tile
  const int ln = lane & 15;
  const int hi8 = lane >> 4;
  const int n_tile = blockIdx.x;           // 0..63
  const int col = n_tile * 16 + ln;
  const int kb_a = hi8 ? 16 : 0;
  const int kb_b = hi8 ? 16 : 0;

  // --- stage W_s slice into LDS with CDNA5 async-to-LDS (tracked by ASYNCcnt) ---
  for (int r = 0; r < 80; ++r) {           // 5 gates x 16 cols
    const int b = r >> 4, c = r & 15;
    const __bf16* src = WsT + (size_t)(b * HH + n_tile * 16 + c) * HH + threadIdx.x * 8;
    unsigned dst = (unsigned)(uintptr_t)(ldsB + r * LDSK + threadIdx.x * 8);
    asm volatile("global_load_async_to_lds_b128 %0, %1, off"
                 :: "v"(dst), "v"(src) : "memory");
  }
  asm volatile("s_wait_asynccnt 0x0" ::: "memory");
  __syncthreads();

  float bsv[5];
#pragma unroll
  for (int b = 0; b < 5; ++b) bsv[b] = b_s[b * HH + col];

  v16bf zfrag;
#pragma unroll
  for (int i = 0; i < 16; ++i) zfrag[i] = (__bf16)0.0f;

  const __bf16* lrow = ldsB + (size_t)ln * LDSK + kb_b;
  const __bf16* hbufs[2] = {hbuf0, hbuf1};
  float* cbufs[2] = {cbuf0, cbuf1};

  for (int t = 0; t < T_STEPS; ++t) {
    const __bf16* h_read = hbufs[t & 1];
    const float*  c_read = cbufs[t & 1];
    __bf16* h_write = (__bf16*)hbufs[(t + 1) & 1];
    float*  c_write = cbufs[(t + 1) & 1];

    // A gather row for this lane (padded rows contribute zeros)
    const int jrow = wave * 16 + ln;
    const __bf16* aptr = nullptr;
    if (jrow < BB) {
      int lbl = labels[t * BB + jrow] % RR;
      aptr = h_read + (size_t)lbl * HH + kb_a;
    }

    v8f acc[5];
#pragma unroll
    for (int b = 0; b < 5; ++b)
#pragma unroll
      for (int i = 0; i < 8; ++i) acc[b][i] = 0.0f;

    for (int k0 = 0; k0 < HH; k0 += 32) {
      v16bf afrag = aptr ? load_a_bf16(aptr + k0) : zfrag;
#pragma unroll
      for (int b = 0; b < 5; ++b) {
        v16bf bfrag = load_b_bf16(lrow + (size_t)(b * 16) * LDSK + k0);  // ds_load path
        acc[b] = __builtin_amdgcn_wmma_f32_16x16x32_bf16(
            false, afrag, false, bfrag, (short)0, acc[b], false, false);
      }
    }

    // fused gates + scatter
#pragma unroll
    for (int i = 0; i < 8; ++i) {
      const int j = wave * 16 + i + 8 * hi8;
      if (j >= BB) continue;
      const size_t grow = (size_t)t * BB + j;
      const float* pinrow = pin_all + grow * N6H;
      const int lblj = labels[grow] % RR;
      const float prevc = c_read[(size_t)lblj * HH + col];

      const float ig = sigmoidf_(pinrow[0 * HH + col] + acc[0][i] + bsv[0]);
      const float fg = sigmoidf_(pinrow[1 * HH + col] + acc[1][i] + bsv[1]);
      const float g  = tanhf   (pinrow[2 * HH + col] + acc[2][i] + bsv[2]);
      const float og = sigmoidf_(pinrow[3 * HH + col] + acc[3][i] + bsv[3]);
      const float c  = ig * g + fg * prevc;
      const float hr = og * tanhf(c);
      const float hw = sigmoidf_(pinrow[4 * HH + col] + acc[4][i] + bsv[4]);
      const float h  = hw * hr + (1.0f - hw) * pinrow[5 * HH + col];

      h_write[(size_t)lblj * HH + col] = (__bf16)h;
      c_write[(size_t)lblj * HH + col] = c;
      HallBF[grow * HH + col] = (__bf16)h;
    }

    // software grid barrier (release -> arrive -> spin -> acquire)
    __threadfence();
    __syncthreads();
    if (threadIdx.x == 0) {
      __hip_atomic_fetch_add(bar, 1, __ATOMIC_RELEASE, __HIP_MEMORY_SCOPE_AGENT);
      const int target = NBLK * (t + 1);
      while (__hip_atomic_load(bar, __ATOMIC_ACQUIRE, __HIP_MEMORY_SCOPE_AGENT) < target) {
        __builtin_amdgcn_s_sleep(1);
      }
    }
    __syncthreads();
    __threadfence();
  }
}

// ---------- final GEMM: outs[M,51] = HallBF[M,H] @ W_out + b_out ----------
__global__ __launch_bounds__(128) void k_gemm_out(
    const __bf16* __restrict__ Hall,   // [MTOT, HH] bf16
    const __bf16* __restrict__ WoT,    // [64, HH] bf16, rows >=51 zero
    const float* __restrict__ b_out,   // [RR]
    float* __restrict__ out)           // [MTOT, RR]
{
  const int lane = threadIdx.x & 31;
  const int wave = threadIdx.x >> 5;
  const int ln = lane & 15;
  const int hi8 = lane >> 4;
  const int m0 = (blockIdx.x * 4 + wave) * 16;

  const __bf16* arow = Hall + (size_t)(m0 + ln) * HH + (hi8 ? 16 : 0);
  const int kb_b = hi8 ? 16 : 0;

  v8f acc[4];
#pragma unroll
  for (int j = 0; j < 4; ++j)
#pragma unroll
    for (int i = 0; i < 8; ++i) acc[j][i] = 0.0f;

  for (int k0 = 0; k0 < HH; k0 += 32) {
    v16bf afrag = load_a_bf16(arow + k0);
#pragma unroll
    for (int j = 0; j < 4; ++j) {
      v16bf bfrag = load_b_bf16(WoT + (size_t)(j * 16 + ln) * HH + k0 + kb_b);
      acc[j] = __builtin_amdgcn_wmma_f32_16x16x32_bf16(
          false, afrag, false, bfrag, (short)0, acc[j], false, false);
    }
  }
#pragma unroll
  for (int j = 0; j < 4; ++j) {
    const int n = j * 16 + ln;
    if (n >= RR) continue;
    const float bb = b_out[n];
#pragma unroll
    for (int i = 0; i < 8; ++i) {
      const int m = m0 + i + 8 * hi8;
      out[(size_t)m * RR + n] = acc[j][i] + bb;
    }
  }
}

static inline size_t align256(size_t x) { return (x + 255) & ~(size_t)255; }

extern "C" void kernel_launch(void* const* d_in, const int* in_sizes, int n_in,
                              void* d_out, int out_size, void* d_ws, size_t ws_size,
                              hipStream_t stream) {
  (void)in_sizes; (void)n_in; (void)out_size; (void)ws_size;
  const float* seq    = (const float*)d_in[0];
  const int*   rel    = (const int*)  d_in[1];
  const float* W_in   = (const float*)d_in[2];
  const float* b_in   = (const float*)d_in[3];
  const float* W_s    = (const float*)d_in[4];
  const float* b_s    = (const float*)d_in[5];
  const float* W_out  = (const float*)d_in[6];
  const float* b_out  = (const float*)d_in[7];
  const float* mem_h0 = (const float*)d_in[8];
  const float* mem_c0 = (const float*)d_in[9];
  float* out = (float*)d_out;

  // workspace layout
  char* base = (char*)d_ws;
  size_t off = 0;
  float*  pin    = (float*) (base + off); off += align256((size_t)MTOT * N6H * 4);
  __bf16* seqBF  = (__bf16*)(base + off); off += align256((size_t)MTOT * DD * 2);
  __bf16* HallBF = (__bf16*)(base + off); off += align256((size_t)MTOT * HH * 2);
  __bf16* WinT   = (__bf16*)(base + off); off += align256((size_t)N6H * DD * 2);
  __bf16* WsT    = (__bf16*)(base + off); off += align256((size_t)N5H * HH * 2);
  __bf16* WoT    = (__bf16*)(base + off); off += align256((size_t)64 * HH * 2);
  __bf16* stH[2]; float* stC[2];
  for (int p = 0; p < 2; ++p) {
    stH[p] = (__bf16*)(base + off); off += align256((size_t)RR * HH * 2);
    stC[p] = (float*) (base + off); off += align256((size_t)RR * HH * 4);
  }
  int* bar = (int*)(base + off); off += 256;

  // prep: stage inputs into WMMA-friendly bf16 layouts + init state/barrier
  k_transpose_cvt<<<4096, 256, 0, stream>>>(W_in, WinT, DD, N6H, (long long)N6H * DD);
  k_transpose_cvt<<<2048, 256, 0, stream>>>(W_s, WsT, HH, N5H, (long long)N5H * HH);
  k_transpose_cvt<<<64,   256, 0, stream>>>(W_out, WoT, HH, RR, (long long)64 * HH);
  k_cvt_bf16_v4<<<4096, 256, 0, stream>>>(seq, seqBF, (long long)MTOT * DD / 4);
  k_cvt_bf16_v4<<<52, 256, 0, stream>>>(mem_h0, stH[0], (long long)RR * HH / 4);
  k_copy_f32<<<204, 256, 0, stream>>>(mem_c0, stC[0], RR * HH);
  k_zero_i32<<<1, 64, 0, stream>>>(bar, 1);

  // big input-projection GEMM (no recurrent dependency): 1.31 TFLOP bf16 WMMA
  {
    dim3 grid(N6H / 128, MTOT / 64);   // (48, 408)
    k_gemm_pin<<<grid, 128, 0, stream>>>(seqBF, WinT, b_in, pin);
  }

  // full 512-step recurrence in ONE persistent launch
  k_recurrence<<<NBLK, 128, 0, stream>>>(pin, WsT, b_s,
                                         stH[0], stC[0], stH[1], stC[1],
                                         HallBF, rel, bar);

  // hoisted output projection for all timesteps
  k_gemm_out<<<MTOT / 64, 128, 0, stream>>>(HallBF, WoT, b_out, out);
}
